// CPCA_34694745817853
// MI455X (gfx1250) — compile-verified, matrix operand order checked by reading, not run on hardware
//
// CPC-action loss for MI455X (gfx1250), wave32, bf16 WMMA.
//
// FLOP budget: GRU hh GEMMs 38.7 GF + MLP L1 135 GF + MLP L2 68 GF  (bf16)
// Byte budget: ~0.5 GB reads, working set < 192MB L2  ->  compute bound
// => everything big runs on v_wmma_f32_16x16x32_bf16, f32 accumulate.
// LDS staging of A tiles uses gfx1250 async global->LDS DMA (ASYNCcnt).
// Loss reduction is fully deterministic (fixed-order partials + tree).

#include <hip/hip_runtime.h>
#include <math.h>

typedef __bf16 bf16_t;
typedef __attribute__((ext_vector_type(16))) __bf16 v16bf;
typedef __attribute__((ext_vector_type(8)))  __bf16 v8bf;
typedef __attribute__((ext_vector_type(8)))  float  v8f;

#define N_    512
#define T_    128
#define H_    512
#define TS_   6
#define FS_   2
#define K_    8
#define A_    17
#define ED_   32
#define NNEG_ 20
#define NR_      (N_*TS_)          // 3072 GRU rows
#define POSROWS  (FS_*NR_)         // 6144
#define NEGROWS  (FS_*NR_*NNEG_)   // 122880
#define TOTROWS  (POSROWS+NEGROWS) // 129024
#define MLPBLK   (TOTROWS/32)      // 4032 workgroups

// ---- WMMA helpers -----------------------------------------------------
// Per-lane fragment layout (ISA 7.12.2, 16-bit A 16x32 / B 32x16):
// lanes 0-15 hold K {kc..kc+7} and {kc+16..kc+23}; lanes 16-31 the other halves.
__device__ __forceinline__ v16bf load_frag(const bf16_t* rowbase, int kc, int kh) {
  union { v16bf v; v8bf h[2]; } u;
  u.h[0] = *(const v8bf*)(rowbase + kc + kh * 8);
  u.h[1] = *(const v8bf*)(rowbase + kc + 16 + kh * 8);
  return u.v;
}
__device__ __forceinline__ v8f wmma_bf16(v16bf a, v16bf b, v8f c) {
  return __builtin_amdgcn_wmma_f32_16x16x32_bf16(false, a, false, b, (short)0, c,
                                                 false, false);
}
__device__ __forceinline__ float sigm(float x) { return 1.0f / (1.0f + expf(-x)); }
__device__ __forceinline__ float softplus(float x) {
  return fmaxf(x, 0.0f) + log1pf(expf(-fabsf(x)));
}

// ---- gfx1250 async global->LDS copy (16B per lane, ASYNCcnt) ----------
// GV mode: global_load_async_to_lds_b128 vdst(lds byte addr), v[addr64], off
// LDS operand = low 32 bits of the shared-aperture generic pointer.
__device__ __forceinline__ void async_copy_b128(void* lds_dst, const void* gsrc) {
  unsigned l = (unsigned)(uintptr_t)lds_dst;
  asm volatile("global_load_async_to_lds_b128 %0, %1, off"
               :: "v"(l), "v"(gsrc) : "memory");
}
__device__ __forceinline__ void async_wait0() {
  asm volatile("s_wait_asynccnt 0x0" ::: "memory");
}

// ---- small prep kernels ----------------------------------------------
__global__ void init_zero(float* denom) {
  if (threadIdx.x == 0) { *denom = 0.0f; }
}

__global__ void cvt_bf16(const float* __restrict__ src, bf16_t* __restrict__ dst, int n) {
  int id = blockIdx.x * 256 + threadIdx.x;
  if (id < n) dst[id] = (bf16_t)src[id];
}

// act_gi[a][j] = embed_w[a] . gru_w_ih[j] + b_ih[j]   (17 x 1536, K=32: tiny)
__global__ void actgi_kernel(const float* __restrict__ embed_w,
                             const float* __restrict__ w_ih,
                             const float* __restrict__ b_ih,
                             float* __restrict__ actgi) {
  int id = blockIdx.x * 256 + threadIdx.x;
  if (id >= A_ * 3 * H_) return;
  int a = id / (3 * H_), j = id % (3 * H_);
  float s = b_ih[j];
#pragma unroll
  for (int e = 0; e < ED_; ++e) s += embed_w[a * ED_ + e] * w_ih[j * ED_ + e];
  actgi[id] = s;
}

// h0[row] = rnn_outputs[ni, time_sub[si], :]
__global__ void h0_kernel(const float* __restrict__ rnn_outputs,
                          const int* __restrict__ time_sub,
                          float* __restrict__ h32, bf16_t* __restrict__ hbf) {
  int id = blockIdx.x * 256 + threadIdx.x;     // NR_*H_
  int row = id >> 9, hc = id & (H_ - 1);
  int ni = row / TS_, si = row % TS_;
  float v = rnn_outputs[((size_t)ni * T_ + time_sub[si]) * H_ + hc];
  h32[id] = v;
  hbf[id] = (bf16_t)v;
}

// mask[f][row] = AND over j=0..u_f of validity at t = ts[si]+j  (OOB => 0)
// denom atomic is a sum of 0/1 floats: exact in any order -> deterministic.
__global__ void mask_kernel(const float* __restrict__ not_dones,
                            const int* __restrict__ valids,
                            const int* __restrict__ time_sub,
                            const int* __restrict__ unroll_sub,
                            float* __restrict__ maskb, float* __restrict__ denom) {
  int id = blockIdx.x * 256 + threadIdx.x;
  if (id >= FS_ * NR_) return;
  int f = id / NR_, row = id % NR_;
  int ni = row / TS_, si = row % TS_;
  int u = unroll_sub[f], tb = time_sub[si];
  float m = 1.0f;
  for (int j = 0; j <= u; ++j) {
    int tk = tb + j;
    bool ok = (tk <= T_ - 2) && (not_dones[ni * T_ + tk] > 0.0f) &&
              (valids[ni * T_ + tk] != 0);
    if (!ok) m = 0.0f;
  }
  maskb[id] = m;
  atomicAdd(denom, m);
}

// ---- GRU hidden GEMM: gh = h @ W_hh^T  (3072 x 1536, K=512) -----------
// grid (96, 6): 32-row x 256-col tile per WG, 8 waves, 2x2 16x16 tiles/wave.
__global__ __launch_bounds__(256) void gh_gemm(const bf16_t* __restrict__ hbf,
                                               const bf16_t* __restrict__ whh,
                                               float* __restrict__ gh) {
  __shared__ bf16_t As[32 * 520];   // padded stride 520 halves (conflict-free)
  int t = threadIdx.x;
  int row0 = blockIdx.x * 32;
  int col0 = blockIdx.y * 256;
  // stage A: 32 rows x 512 bf16 = 2048 chunks of 8 halves, async DMA to LDS
  for (int i = 0; i < 8; ++i) {
    int idx = i * 256 + t;
    int r = idx >> 6, c8 = (idx & 63) * 8;
    async_copy_b128(&As[r * 520 + c8], hbf + (size_t)(row0 + r) * H_ + c8);
  }
  async_wait0();
  __syncthreads();
  int wave = t >> 5, lane = t & 31;
  int ln = lane & 15, kh = lane >> 4;
  for (int cc = 0; cc < 2; ++cc) {
    int ct = wave * 2 + cc;
    int jcol = col0 + ct * 16 + ln;               // W_hh row = B column
    const bf16_t* brow = whh + (size_t)jcol * H_;
    v8f acc0 = {}, acc1 = {};
    for (int kc = 0; kc < H_; kc += 32) {
      v16bf b  = load_frag(brow, kc, kh);
      v16bf a0 = load_frag(&As[(0 * 16 + ln) * 520], kc, kh);
      v16bf a1 = load_frag(&As[(1 * 16 + ln) * 520], kc, kh);
      acc0 = wmma_bf16(a0, b, acc0);
      acc1 = wmma_bf16(a1, b, acc1);
    }
#pragma unroll
    for (int r = 0; r < 8; ++r) {
      int n = col0 + ct * 16 + ln;
      gh[(size_t)(row0 + r + kh * 8) * 1536 + n]      = acc0[r];
      gh[(size_t)(row0 + 16 + r + kh * 8) * 1536 + n] = acc1[r];
    }
  }
}

// ---- GRU gate + preds snapshot ---------------------------------------
__global__ void gru_gate(const float* __restrict__ gh, const float* __restrict__ actgi,
                         const float* __restrict__ b_ih, const float* __restrict__ b_hh,
                         const int* __restrict__ actions, const int* __restrict__ time_sub,
                         const int* __restrict__ unroll_sub,
                         float* __restrict__ h32, bf16_t* __restrict__ hbf,
                         bf16_t* __restrict__ fpbf, int k) {
  int id = blockIdx.x * 256 + threadIdx.x;       // NR_*H_
  int row = id >> 9, hc = id & (H_ - 1);
  int ni = row / TS_, si = row % TS_;
  int tk = time_sub[si] + k;
  float gir, giz, gin;
  if (tk <= T_ - 2) {
    int a = actions[ni * T_ + tk];
    const float* g = actgi + (size_t)a * 1536;
    gir = g[hc]; giz = g[hc + H_]; gin = g[hc + 2 * H_];
  } else {                                        // zero embedding -> bias only
    gir = b_ih[hc]; giz = b_ih[hc + H_]; gin = b_ih[hc + 2 * H_];
  }
  const float* ghr = gh + (size_t)row * 1536;
  float hr = ghr[hc] + b_hh[hc];
  float hz = ghr[hc + H_] + b_hh[hc + H_];
  float hn = ghr[hc + 2 * H_] + b_hh[hc + 2 * H_];
  float r = sigm(gir + hr);
  float z = sigm(giz + hz);
  float c = tanhf(gin + r * hn);
  float hnew = (1.0f - z) * c + z * h32[id];
  h32[id] = hnew;
  hbf[id] = (bf16_t)hnew;
  if (unroll_sub[0] == k) fpbf[id] = (bf16_t)hnew;
  if (unroll_sub[1] == k) fpbf[(size_t)NR_ * H_ + id] = (bf16_t)hnew;
}

// ---- fused 3-layer MLP + softplus loss -------------------------------
// 32 output rows per WG.  Dyn LDS: A 32x1040 halves, Y1 32x520 halves (~100KB).
// Emits one fixed-order partial per workgroup (deterministic).
__global__ __launch_bounds__(256) void mlp_kernel(
    const bf16_t* __restrict__ fpbf, const float* __restrict__ rnn_inputs,
    const bf16_t* __restrict__ w1, const bf16_t* __restrict__ w2,
    const float* __restrict__ b1, const float* __restrict__ b2,
    const float* __restrict__ w3, const float* __restrict__ b3,
    const float* __restrict__ maskb, const float* __restrict__ denomp,
    const int* __restrict__ time_sub, const int* __restrict__ unroll_sub,
    const int* __restrict__ neg_idx, float* __restrict__ partials) {
  extern __shared__ char smem_raw[];
  bf16_t* As  = (bf16_t*)smem_raw;       // [32][1040]  (K=1024 + pad 16)
  bf16_t* Y1s = As + 32 * 1040;          // [32][520]
  bf16_t* Y2s = As;                      // reuse A space after layer1
  __shared__ int   s_fprow[32];
  __shared__ int   s_feat[32];
  __shared__ float s_w[32];
  __shared__ float s_sgn[32];
  __shared__ float s_contrib[32];

  int t = threadIdx.x;
  int r0 = blockIdx.x * 32;
  if (t < 32) {                           // per-row spec
    int r = r0 + t;
    float denom = *denomp;
    int f, row, featRow; float sgn, w;
    if (r < POSROWS) {                    // positive: feature = future rnn_input
      f = r / NR_; row = r % NR_;
      int ni = row / TS_, si = row % TS_;
      int tt = time_sub[si] + unroll_sub[f];
      featRow = (tt <= T_ - 2) ? (ni * T_ + tt + 1) : -1;   // OOB -> zeros
      sgn = -1.0f;                        // softplus(-logit)
      w = maskb[f * NR_ + row] * 0.1f / denom;
    } else {                              // negative: gathered random feature
      int rr = r - POSROWS;
      f = rr / (NR_ * NNEG_);
      int rem = rr % (NR_ * NNEG_);
      row = rem / NNEG_;
      int j = rem % NNEG_;
      featRow = neg_idx[((size_t)f * NR_ + row) * NNEG_ + j];
      sgn = 1.0f;                         // softplus(+logit)
      w = maskb[f * NR_ + row] * 0.1f / (denom * (float)NNEG_);
    }
    s_fprow[t] = f * NR_ + row;
    s_feat[t]  = featRow;
    s_w[t]     = w;
    s_sgn[t]   = sgn;
  }
  __syncthreads();

  // stage A left half (fp, bf16): 32 x 512 halves, async DMA to LDS
  for (int i = 0; i < 8; ++i) {
    int idx = i * 256 + t;
    int rr = idx >> 6, c8 = (idx & 63) * 8;
    async_copy_b128(As + rr * 1040 + c8, fpbf + (size_t)s_fprow[rr] * H_ + c8);
  }
  // stage A right half (feature, f32 -> bf16 converts through VGPRs)
  for (int i = 0; i < 8; ++i) {
    int idx = i * 256 + t;
    int rr = idx >> 6, c8 = (idx & 63) * 8;
    int fr = s_feat[rr];
    v8bf d;
    if (fr >= 0) {
      const float* src = rnn_inputs + (size_t)fr * H_ + c8;
#pragma unroll
      for (int e = 0; e < 8; ++e) d[e] = (bf16_t)src[e];
    } else {
#pragma unroll
      for (int e = 0; e < 8; ++e) d[e] = (bf16_t)0.0f;
    }
    *(v8bf*)(As + rr * 1040 + 512 + c8) = d;
  }
  async_wait0();
  __syncthreads();

  int wave = t >> 5, lane = t & 31;
  int ln = lane & 15, kh = lane >> 4;

  // Layer 1: relu([32,1024] @ W1^T + b1) -> Y1s [32,512]
  for (int cc = 0; cc < 4; ++cc) {
    int j = (wave * 4 + cc) * 16 + ln;
    const bf16_t* brow = w1 + (size_t)j * 1024;
    v8f acc0 = {}, acc1 = {};
    for (int kc = 0; kc < 1024; kc += 32) {
      v16bf b  = load_frag(brow, kc, kh);
      v16bf a0 = load_frag(As + (0 * 16 + ln) * 1040, kc, kh);
      v16bf a1 = load_frag(As + (1 * 16 + ln) * 1040, kc, kh);
      acc0 = wmma_bf16(a0, b, acc0);
      acc1 = wmma_bf16(a1, b, acc1);
    }
    float bj = b1[j];
#pragma unroll
    for (int r = 0; r < 8; ++r) {
      Y1s[(r + kh * 8) * 520 + j]      = (bf16_t)fmaxf(acc0[r] + bj, 0.0f);
      Y1s[(16 + r + kh * 8) * 520 + j] = (bf16_t)fmaxf(acc1[r] + bj, 0.0f);
    }
  }
  __syncthreads();

  // Layer 2: relu(Y1 @ W2^T + b2) -> Y2s [32,512] (reusing A's LDS)
  for (int cc = 0; cc < 4; ++cc) {
    int j = (wave * 4 + cc) * 16 + ln;
    const bf16_t* brow = w2 + (size_t)j * H_;
    v8f acc0 = {}, acc1 = {};
    for (int kc = 0; kc < H_; kc += 32) {
      v16bf b  = load_frag(brow, kc, kh);
      v16bf a0 = load_frag(Y1s + (0 * 16 + ln) * 520, kc, kh);
      v16bf a1 = load_frag(Y1s + (1 * 16 + ln) * 520, kc, kh);
      acc0 = wmma_bf16(a0, b, acc0);
      acc1 = wmma_bf16(a1, b, acc1);
    }
    float bj = b2[j];
#pragma unroll
    for (int r = 0; r < 8; ++r) {
      Y2s[(r + kh * 8) * 520 + j]      = (bf16_t)fmaxf(acc0[r] + bj, 0.0f);
      Y2s[(16 + r + kh * 8) * 520 + j] = (bf16_t)fmaxf(acc1[r] + bj, 0.0f);
    }
  }
  __syncthreads();

  // Layer 3: logit = Y2 . w3 + b3; masked softplus, per-row contribution
  for (int q = 0; q < 4; ++q) {
    int m = wave * 4 + q;
    float s = 0.0f;
    for (int jj = lane; jj < H_; jj += 32)
      s += (float)Y2s[m * 520 + jj] * w3[jj];
#pragma unroll
    for (int off = 16; off > 0; off >>= 1) s += __shfl_down(s, off, 32);
    if (lane == 0) {
      float logit = s + b3[0];
      s_contrib[m] = s_w[m] * softplus(s_sgn[m] * logit);
    }
  }
  __syncthreads();
  if (t == 0) {                          // fixed-order sum -> deterministic
    float s = 0.0f;
#pragma unroll
    for (int m = 0; m < 32; ++m) s += s_contrib[m];
    partials[blockIdx.x] = s;
  }
}

// ---- deterministic final reduction -----------------------------------
__global__ __launch_bounds__(256) void reduce_partials(const float* __restrict__ parts,
                                                       int n, float* __restrict__ out) {
  __shared__ float buf[256];
  int t = threadIdx.x;
  float s = 0.0f;
  for (int i = t; i < n; i += 256) s += parts[i];   // fixed per-thread order
  buf[t] = s;
  __syncthreads();
  for (int off = 128; off > 0; off >>= 1) {         // fixed tree order
    if (t < off) buf[t] += buf[t + off];
    __syncthreads();
  }
  if (t == 0) *out = buf[0];
}

// ---- launch ----------------------------------------------------------
extern "C" void kernel_launch(void* const* d_in, const int* in_sizes, int n_in,
                              void* d_out, int out_size, void* d_ws, size_t ws_size,
                              hipStream_t stream) {
  const int*   actions     = (const int*)  d_in[0];
  const float* not_dones   = (const float*)d_in[1];
  const int*   valids      = (const int*)  d_in[2];
  const float* rnn_inputs  = (const float*)d_in[3];
  const float* rnn_outputs = (const float*)d_in[4];
  const float* embed_w     = (const float*)d_in[5];
  const float* gru_w_ih    = (const float*)d_in[6];
  const float* gru_w_hh    = (const float*)d_in[7];
  const float* gru_b_ih    = (const float*)d_in[8];
  const float* gru_b_hh    = (const float*)d_in[9];
  const float* p_w1        = (const float*)d_in[10];
  const float* p_b1        = (const float*)d_in[11];
  const float* p_w2        = (const float*)d_in[12];
  const float* p_b2        = (const float*)d_in[13];
  const float* p_w3        = (const float*)d_in[14];
  const float* p_b3        = (const float*)d_in[15];
  const int*   time_sub    = (const int*)  d_in[16];
  const int*   unroll_sub  = (const int*)  d_in[17];
  const int*   neg_idx     = (const int*)  d_in[18];
  float* out = (float*)d_out;

  // workspace carve (~37.9 MB total)
  char* w = (char*)d_ws;
  bf16_t* whh_bf  = (bf16_t*)(w + 0);          // 1536*512*2  = 1572864
  bf16_t* w1_bf   = (bf16_t*)(w + 1572864);    // 512*1024*2  = 1048576
  bf16_t* w2_bf   = (bf16_t*)(w + 2621440);    // 512*512*2   =  524288
  float*  actgi   = (float*) (w + 3145728);    // 17*1536*4   =  104448
  float*  h32     = (float*) (w + 3250176);    // 3072*512*4  = 6291456
  bf16_t* hbf     = (bf16_t*)(w + 9541632);    // 3072*512*2  = 3145728
  float*  gh      = (float*) (w + 12687360);   // 3072*1536*4 = 18874368
  bf16_t* fpbf    = (bf16_t*)(w + 31561728);   // 2*3072*512*2= 6291456
  float*  maskb   = (float*) (w + 37853184);   // 6144*4      =   24576
  float*  denom   = (float*) (w + 37877760);   // 4
  float*  partial = (float*) (w + 37878016);   // 4032*4      =   16128

  init_zero<<<1, 64, 0, stream>>>(denom);
  cvt_bf16<<<(1536 * 512 + 255) / 256, 256, 0, stream>>>(gru_w_hh, whh_bf, 1536 * 512);
  cvt_bf16<<<(512 * 1024 + 255) / 256, 256, 0, stream>>>(p_w1, w1_bf, 512 * 1024);
  cvt_bf16<<<(512 * 512 + 255) / 256, 256, 0, stream>>>(p_w2, w2_bf, 512 * 512);
  actgi_kernel<<<(A_ * 3 * H_ + 255) / 256, 256, 0, stream>>>(embed_w, gru_w_ih,
                                                              gru_b_ih, actgi);
  h0_kernel<<<NR_ * H_ / 256, 256, 0, stream>>>(rnn_outputs, time_sub, h32, hbf);
  mask_kernel<<<(FS_ * NR_ + 255) / 256, 256, 0, stream>>>(not_dones, valids, time_sub,
                                                           unroll_sub, maskb, denom);
  for (int k = 0; k < K_; ++k) {
    dim3 g(NR_ / 32, 1536 / 256);
    gh_gemm<<<g, 256, 0, stream>>>(hbf, whh_bf, gh);
    gru_gate<<<NR_ * H_ / 256, 256, 0, stream>>>(gh, actgi, gru_b_ih, gru_b_hh, actions,
                                                 time_sub, unroll_sub, h32, hbf, fpbf, k);
  }
  size_t mlp_lds = (size_t)(32 * 1040 + 32 * 520) * sizeof(bf16_t);   // 99840 B
  mlp_kernel<<<MLPBLK, 256, mlp_lds, stream>>>(fpbf, rnn_inputs, w1_bf, w2_bf,
                                               p_b1, p_b2, p_w3, p_b3, maskb,
                                               denom, time_sub, unroll_sub,
                                               neg_idx, partial);
  reduce_partials<<<1, 256, 0, stream>>>(partial, MLPBLK, out);
}
